// LatentSDE_71425306132720
// MI455X (gfx1250) — matrix operand order repaired and checked by reading, not compiled, not run on hardware
//
#include <hip/hip_runtime.h>
#include <hip/hip_bf16.h>
#include <math.h>

typedef __attribute__((ext_vector_type(16))) _Float16 v16h;
typedef __attribute__((ext_vector_type(8)))  _Float16 v8h;
typedef __attribute__((ext_vector_type(4)))  _Float16 v4h;
typedef __attribute__((ext_vector_type(8)))  float    v8f;

static constexpr int kT = 512;   // time steps
static constexpr int kB = 512;   // batch
static constexpr int kL = 16;    // latent
static constexpr int kC = 64;    // ctx
static constexpr int kH = 128;   // hidden

// LDS leading dims, padded to kill bank conflicts (64 banks x 4B)
static constexpr int LD_H   = 136;  // f16 rows of 128 (+8)
static constexpr int LD_X   = 40;   // f16 rows of 32  (+8)
static constexpr int LD_IN  = 104;  // f16 rows of 96  (+8)
static constexpr int LD_GS  = 385;  // f32 rows of 384 (+1)
static constexpr int LD_HN  = 129;  // f32 rows of 128 (+1)
static constexpr int LD_S   = 17;   // f32 rows of 16  (+1)
static constexpr int LD_PX  = 33;   // f32 rows of 32  (+1)

#define ACC_LPX  0
#define ACC_LAND 1
#define ACC_MOVE 2
#define ACC_PICK 3
#define ACC_MASK 4
#define ACC_KL   5
#define ACC_LOGR 6
#define ACC_N    16

// ---------------- WMMA helpers (CDNA5 16x16x32 f16 -> f32) ----------------

__device__ __forceinline__ v8f wmma_f16(v16h a, v16h b, v8f c) {
  return __builtin_amdgcn_wmma_f32_16x16x32_f16(false, a, false, b, (short)0, c, false, false);
}

__device__ __forceinline__ v16h cat8(v8h lo, v8h hi) {
  return __builtin_shufflevector(lo, hi, 0, 1, 2, 3, 4, 5, 6, 7, 8, 9, 10, 11, 12, 13, 14, 15);
}

// A fragment: 16x32 f16, row-major source with leading dim ld (rows 16B-aligned).
__device__ __forceinline__ v16h load_a_frag(const _Float16* src, int ld, int lane) {
  int m  = lane & 15;
  int kb = (lane >> 4) << 3;
  const _Float16* p = src + (size_t)m * ld + kb;
  v8h lo = *(const v8h*)p;
  v8h hi = *(const v8h*)(p + 16);
  return cat8(lo, hi);
}

// B fragment, fragment-major: tile*512 + lane*16 contiguous halves (32B aligned)
__device__ __forceinline__ v16h load_b_fr(const _Float16* fr, int tile, int lane) {
  return *(const v16h*)(fr + ((size_t)tile << 9) + ((size_t)lane << 4));
}

// D store to f32 buffer, leading dim ld: D[v + 8*(lane>=16)][lane&15]
__device__ __forceinline__ void store_d(float* dst, int ld, int lane, v8f c) {
  int n  = lane & 15;
  int mb = (lane >> 4) << 3;
#pragma unroll
  for (int v = 0; v < 8; ++v) dst[(size_t)(mb + v) * ld + n] = c[v];
}

__device__ __forceinline__ float softplusf(float x) {
  return (x > 20.f) ? x : __logf(1.f + __expf(x));
}
__device__ __forceinline__ float sigmf(float x) { return 1.f / (1.f + __expf(-x)); }

// ---------------- prep kernels ----------------

__global__ void k_zero_acc(float* acc) {
  int i = threadIdx.x;
  if (i < ACC_N) acc[i] = 0.f;
}

// W: n x kdim row-major (out x in). FR: fragment-major B layout, K padded to kpad.
__global__ void k_fragify(const float* __restrict__ W, _Float16* __restrict__ FR,
                          int n, int kdim, int kpad) {
  int idx = blockIdx.x * blockDim.x + threadIdx.x;
  int ntiles = n >> 4;
  int total = (kpad >> 5) * ntiles * 512;
  if (idx >= total) return;
  int i = idx & 15;
  int lane = (idx >> 4) & 31;
  int tile = idx >> 9;
  int ktile = tile / ntiles, ntile = tile - ktile * ntiles;
  int k = ktile * 32 + ((lane >> 4) << 3) + (i & 7) + ((i >> 3) << 4);
  int col = ntile * 16 + (lane & 15);
  float v = (k < kdim) ? W[(size_t)col * kdim + k] : 0.f;
  FR[idx] = (_Float16)v;
}

// fused head (land|shot|move) fragment-major: K=128 (4 ktiles), N=16
__global__ void k_build_head_fr(const float* __restrict__ landw, const float* __restrict__ shotw,
                                const float* __restrict__ movew, _Float16* __restrict__ FR) {
  int idx = blockIdx.x * blockDim.x + threadIdx.x;
  if (idx >= 4 * 512) return;
  int i = idx & 15;
  int lane = (idx >> 4) & 31;
  int ktile = idx >> 9;
  int k = ktile * 32 + ((lane >> 4) << 3) + (i & 7) + ((i >> 3) << 4);
  int col = lane & 15;
  float v;
  if (col < 2)       v = landw[(size_t)col * 128 + k];
  else if (col < 14) v = shotw[(size_t)(col - 2) * 128 + k];
  else               v = movew[(size_t)(col - 14) * 128 + k];
  FR[idx] = (_Float16)v;
}

// build state_t (embedded 32-dim input) as f16
__global__ void k_embed(const float* __restrict__ xs, const float* __restrict__ shot_emb,
                        const float* __restrict__ player_emb, _Float16* __restrict__ st16) {
  size_t idx = (size_t)blockIdx.x * blockDim.x + threadIdx.x;
  size_t total = (size_t)kT * kB * 32;
  if (idx >= total) return;
  size_t tb = idx >> 5;
  int d = (int)(idx & 31);
  const float* x = xs + tb * 23;
  float v;
  if (d < 12)      v = x[d];
  else if (d < 20) v = shot_emb[((int)x[12]) * 8 + (d - 12)];
  else if (d < 24) v = x[13 + (d - 20)];
  else             v = player_emb[((int)x[17]) * 8 + (d - 24)];
  st16[idx] = (_Float16)v;
}

// ---------------- GRU (backward scan) + encoder ----------------

__global__ void __launch_bounds__(256)
k_gru(const _Float16* __restrict__ st16, const _Float16* __restrict__ WihF,
      const _Float16* __restrict__ WhhF, const _Float16* __restrict__ EncF,
      const float* __restrict__ bih, const float* __restrict__ bhh,
      const float* __restrict__ enc_b, _Float16* __restrict__ ctx16) {
  __shared__ __align__(32) _Float16 hbuf[16 * LD_H];
  __shared__ __align__(32) _Float16 xbuf[16 * LD_X];
  __shared__ float gsum[16 * LD_GS];
  __shared__ float hnb[16 * LD_HN];
  __shared__ float bihS[384], bhhS[384], encbS[64];

  const int tid  = threadIdx.x;
  const int lane = tid & 31;
  const int w    = tid >> 5;
  const int b0   = blockIdx.x * 16;

  for (int i = tid; i < 16 * LD_H; i += 256) hbuf[i] = (_Float16)0.f;
  for (int i = tid; i < 384; i += 256) { bihS[i] = bih[i]; bhhS[i] = bhh[i]; }
  if (tid < 64) encbS[tid] = enc_b[tid];

  // hoist loop-invariant weight fragments into registers
  v16h wih_f[3], whh_f[3][4];
#pragma unroll
  for (int tt = 0; tt < 3; ++tt) {
    int tile = w * 3 + tt;
    wih_f[tt] = load_b_fr(WihF, tile, lane);
#pragma unroll
    for (int kk = 0; kk < 4; ++kk) whh_f[tt][kk] = load_b_fr(WhhF, kk * 24 + tile, lane);
  }
  __syncthreads();

  for (int t = kT - 1; t >= 0; --t) {
    // stage x tile (8B vector copies: 128 threads x 4 halves)
    if (tid < 128) {
      int r = tid >> 3, c4 = (tid & 7) << 2;
      *(v4h*)(xbuf + r * LD_X + c4) =
          *(const v4h*)(st16 + ((size_t)t * kB + b0 + r) * 32 + c4);
    }
    // prefetch next step's x tile into near cache while this step computes
    if (t > 0 && tid < 16)
      __builtin_prefetch(st16 + ((size_t)(t - 1) * kB + b0 + tid) * 32, 0, 3);
    __syncthreads();

    v16h ax = load_a_frag(xbuf, LD_X, lane);
    v16h ah[4];
#pragma unroll
    for (int kk = 0; kk < 4; ++kk) ah[kk] = load_a_frag(hbuf + kk * 32, LD_H, lane);

#pragma unroll
    for (int tt = 0; tt < 3; ++tt) {
      int n0 = (w * 3 + tt) * 16;
      v8f acc = {};
      acc = wmma_f16(ax, wih_f[tt], acc);
      if (n0 < 256) {  // r,z gates: combined x+h sums
#pragma unroll
        for (int kk = 0; kk < 4; ++kk) acc = wmma_f16(ah[kk], whh_f[tt][kk], acc);
        store_d(gsum + n0, LD_GS, lane, acc);
      } else {         // n gate: x part and h part kept separate
        store_d(gsum + n0, LD_GS, lane, acc);
        v8f acc2 = {};
#pragma unroll
        for (int kk = 0; kk < 4; ++kk) acc2 = wmma_f16(ah[kk], whh_f[tt][kk], acc2);
        store_d(hnb + (n0 - 256), LD_HN, lane, acc2);
      }
    }
    __syncthreads();

    // gate nonlinearity + state update (fp32 VALU)
    for (int i = tid; i < 16 * 128; i += 256) {
      int r = i >> 7, c = i & 127;
      float rs = gsum[r * LD_GS + c]       + bihS[c]       + bhhS[c];
      float zs = gsum[r * LD_GS + 128 + c] + bihS[128 + c] + bhhS[128 + c];
      float in = gsum[r * LD_GS + 256 + c] + bihS[256 + c];
      float hn = hnb[r * LD_HN + c]        + bhhS[256 + c];
      float rr = sigmf(rs);
      float zz = sigmf(zs);
      float nn = tanhf(in + rr * hn);
      float ho = (float)hbuf[r * LD_H + c];
      hbuf[r * LD_H + c] = (_Float16)((1.f - zz) * nn + zz * ho);
    }
    __syncthreads();

    // ctx[t] = h @ enc + enc_b (waves 0..3)
    if (w < 4) {
      v8f acc = {};
#pragma unroll
      for (int kk = 0; kk < 4; ++kk)
        acc = wmma_f16(load_a_frag(hbuf + kk * 32, LD_H, lane),
                       load_b_fr(EncF, kk * 4 + w, lane), acc);
      int n = lane & 15, mb = (lane >> 4) << 3;
#pragma unroll
      for (int v = 0; v < 8; ++v) {
        int col = w * 16 + n;
        ctx16[((size_t)t * kB + b0 + mb + v) * 64 + col] = (_Float16)(acc[v] + encbS[col]);
      }
    }
    __syncthreads();
  }
}

// ---------------- qz0 / z0 / KL ----------------

__global__ void k_qz0(const _Float16* __restrict__ ctx16, const float* __restrict__ qz0_w,
                      const float* __restrict__ qz0_b, const float* __restrict__ z0_noise,
                      const float* __restrict__ pz0_mean, const float* __restrict__ pz0_logstd,
                      float* __restrict__ zs, float* __restrict__ acc) {
  int b = blockIdx.x * blockDim.x + threadIdx.x;
  if (b >= kB) return;
  float c[64];
  for (int k = 0; k < 64; ++k) c[k] = (float)ctx16[(size_t)b * 64 + k];  // ctx[0]
  float klsum = 0.f;
  for (int l = 0; l < 16; ++l) {
    float qm = qz0_b[l], ql = qz0_b[16 + l];
    for (int k = 0; k < 64; ++k) {
      qm += qz0_w[(size_t)l * 64 + k] * c[k];
      ql += qz0_w[(size_t)(16 + l) * 64 + k] * c[k];
    }
    float z0 = qm + __expf(ql) * z0_noise[(size_t)b * 16 + l];
    zs[(size_t)b * 16 + l] = z0;
    float vq = __expf(2.f * ql);
    float vp = __expf(2.f * pz0_logstd[l]);
    float dm = qm - pz0_mean[l];
    klsum += pz0_logstd[l] - ql + (vq + dm * dm) / (2.f * vp) - 0.5f;
  }
  atomicAdd(&acc[ACC_KL], klsum);
}

// ---------------- SDE forward scan ----------------

__global__ void __launch_bounds__(256)
k_sde(const _Float16* __restrict__ ctx16, const float* __restrict__ ts,
      const float* __restrict__ bm_noise,
      const _Float16* __restrict__ F1F, const float* __restrict__ f_b1,
      const _Float16* __restrict__ F2F, const float* __restrict__ f_b2,
      const _Float16* __restrict__ F3F, const float* __restrict__ f_b3,
      const _Float16* __restrict__ H1F, const float* __restrict__ h_b1,
      const _Float16* __restrict__ H2F, const float* __restrict__ h_b2,
      const _Float16* __restrict__ H3F, const float* __restrict__ h_b3,
      const float* __restrict__ g_w1, const float* __restrict__ g_b1,
      const float* __restrict__ g_w2, const float* __restrict__ g_b2,
      float* __restrict__ zs, float* __restrict__ acc) {
  __shared__ __align__(32) _Float16 inbuf[16 * LD_IN];  // [z(16) | c(64) | pad]
  __shared__ __align__(32) _Float16 zh[16 * LD_X];
  __shared__ __align__(32) _Float16 m1[16 * LD_H];
  __shared__ __align__(32) _Float16 m2[16 * LD_H];
  __shared__ float fv[16 * LD_S];
  __shared__ float hv[16 * LD_S];
  __shared__ float gv[16 * LD_S];
  __shared__ float zcur[16 * 16];
  __shared__ float red[256];
  __shared__ float gw1S[2048], gb1S[2048], gw2S[2048], gb2S[16];
  __shared__ float fb1S[128], fb2S[128], fb3S[16], hb1S[128], hb2S[128], hb3S[16];

  const int tid = threadIdx.x;
  const int lane = tid & 31;
  const int w = tid >> 5;
  const int b0 = blockIdx.x * 16;
  const int r = tid >> 4, l = tid & 15;

  for (int i = tid; i < 2048; i += 256) {
    gw1S[i] = g_w1[i]; gb1S[i] = g_b1[i]; gw2S[i] = g_w2[i];
  }
  if (tid < 128) { fb1S[tid] = f_b1[tid]; fb2S[tid] = f_b2[tid];
                   hb1S[tid] = h_b1[tid]; hb2S[tid] = h_b2[tid]; }
  if (tid < 16) { fb3S[tid] = f_b3[tid]; hb3S[tid] = h_b3[tid]; gb2S[tid] = g_b2[tid]; }

  // hoist loop-invariant fragments (per-wave tile = w)
  v16h f1_f[3], f2_f[4], h2_f[4], h1_f;
#pragma unroll
  for (int kk = 0; kk < 3; ++kk) f1_f[kk] = load_b_fr(F1F, kk * 8 + w, lane);
#pragma unroll
  for (int kk = 0; kk < 4; ++kk) { f2_f[kk] = load_b_fr(F2F, kk * 8 + w, lane);
                                   h2_f[kk] = load_b_fr(H2F, kk * 8 + w, lane); }
  h1_f = load_b_fr(H1F, w, lane);

  zcur[tid] = zs[(size_t)(b0 + r) * 16 + l];  // z0 slab
  float logr_acc = 0.f;
  __syncthreads();

  for (int t = 0; t < kT - 1; ++t) {
    float dt = ts[t + 1] - ts[t];
    float sdt = sqrtf(dt);

    // stage inputs + diagonal diffusion g(z)
    {
      float z = zcur[r * 16 + l];
      inbuf[r * LD_IN + l] = (_Float16)z;
      zh[r * LD_X + l] = (_Float16)z;
      zh[r * LD_X + 16 + l] = (_Float16)0.f;
      *(v4h*)(inbuf + r * LD_IN + 16 + l * 4) =
          *(const v4h*)(ctx16 + ((size_t)(t + 1) * kB + b0 + r) * 64 + l * 4);
      inbuf[r * LD_IN + 80 + l] = (_Float16)0.f;
      // prefetch next step's ctx + noise rows into near cache
      if (l == 0 && t + 2 < kT)
        __builtin_prefetch(ctx16 + ((size_t)(t + 2) * kB + b0 + r) * 64, 0, 3);
      if (l == 1 && t + 1 < kT - 1)
        __builtin_prefetch(bm_noise + ((size_t)(t + 1) * kB + b0 + r) * 16, 0, 3);
      float s = gb2S[l];
      for (int hh = 0; hh < 128; ++hh) {
        float pre = z * gw1S[l * 128 + hh] + gb1S[l * 128 + hh];
        s += softplusf(pre) * gw2S[l * 128 + hh];
      }
      gv[r * LD_S + l] = sigmf(s);
    }
    __syncthreads();

    // f layer 1: 96 -> 128
    {
      v8f a = {};
#pragma unroll
      for (int kk = 0; kk < 3; ++kk)
        a = wmma_f16(load_a_frag(inbuf + kk * 32, LD_IN, lane), f1_f[kk], a);
      int n = lane & 15, mb = (lane >> 4) << 3;
#pragma unroll
      for (int v = 0; v < 8; ++v)
        m1[(mb + v) * LD_H + w * 16 + n] = (_Float16)softplusf(a[v] + fb1S[w * 16 + n]);
    }
    __syncthreads();

    // f layer 2: 128 -> 128
    {
      v8f a = {};
#pragma unroll
      for (int kk = 0; kk < 4; ++kk)
        a = wmma_f16(load_a_frag(m1 + kk * 32, LD_H, lane), f2_f[kk], a);
      int n = lane & 15, mb = (lane >> 4) << 3;
#pragma unroll
      for (int v = 0; v < 8; ++v)
        m2[(mb + v) * LD_H + w * 16 + n] = (_Float16)softplusf(a[v] + fb2S[w * 16 + n]);
    }
    __syncthreads();

    // h layer 1 (all waves); wave 0 also f layer 3 (m2 -> fv)
    {
      v8f a = {};
      a = wmma_f16(load_a_frag(zh, LD_X, lane), h1_f, a);
      int n = lane & 15, mb = (lane >> 4) << 3;
#pragma unroll
      for (int v = 0; v < 8; ++v)
        m1[(mb + v) * LD_H + w * 16 + n] = (_Float16)softplusf(a[v] + hb1S[w * 16 + n]);
      if (w == 0) {
        v8f f = {};
#pragma unroll
        for (int kk = 0; kk < 4; ++kk)
          f = wmma_f16(load_a_frag(m2 + kk * 32, LD_H, lane), load_b_fr(F3F, kk, lane), f);
#pragma unroll
        for (int v = 0; v < 8; ++v) fv[(mb + v) * LD_S + n] = f[v] + fb3S[n];
      }
    }
    __syncthreads();

    // h layer 2: m1 -> m2
    {
      v8f a = {};
#pragma unroll
      for (int kk = 0; kk < 4; ++kk)
        a = wmma_f16(load_a_frag(m1 + kk * 32, LD_H, lane), h2_f[kk], a);
      int n = lane & 15, mb = (lane >> 4) << 3;
#pragma unroll
      for (int v = 0; v < 8; ++v)
        m2[(mb + v) * LD_H + w * 16 + n] = (_Float16)softplusf(a[v] + hb2S[w * 16 + n]);
    }
    __syncthreads();

    // h layer 3: m2 -> hv (wave 0)
    if (w == 0) {
      v8f h = {};
#pragma unroll
      for (int kk = 0; kk < 4; ++kk)
        h = wmma_f16(load_a_frag(m2 + kk * 32, LD_H, lane), load_b_fr(H3F, kk, lane), h);
      int n = lane & 15, mb = (lane >> 4) << 3;
#pragma unroll
      for (int v = 0; v < 8; ++v) hv[(mb + v) * LD_S + n] = h[v] + hb3S[n];
    }
    __syncthreads();

    // Euler-Maruyama update + log-ratio
    {
      float f_ = fv[r * LD_S + l], h_ = hv[r * LD_S + l], g_ = gv[r * LD_S + l];
      float u = (f_ - h_) / g_;
      logr_acc += 0.5f * u * u * dt;
      float dw = sdt * bm_noise[((size_t)t * kB + b0 + r) * 16 + l];
      float zn = zcur[r * 16 + l] + f_ * dt + g_ * dw;
      zcur[r * 16 + l] = zn;
      zs[((size_t)(t + 1) * kB + b0 + r) * 16 + l] = zn;
    }
    __syncthreads();
  }

  red[tid] = logr_acc;
  __syncthreads();
  for (int s = 128; s > 0; s >>= 1) {
    if (tid < s) red[tid] += red[tid + s];
    __syncthreads();
  }
  if (tid == 0) atomicAdd(&acc[ACC_LOGR], red[0]);
}

// ---------------- decode heads + losses ----------------

__global__ void __launch_bounds__(256)
k_decode(const float* __restrict__ zs, const float* __restrict__ xs,
         const float* __restrict__ shot_emb, const float* __restrict__ player_emb,
         const _Float16* __restrict__ ActF, const float* __restrict__ act_b,
         const _Float16* __restrict__ HeadF, const _Float16* __restrict__ ProjF,
         const float* __restrict__ proj_b, const float* __restrict__ noise_std_p,
         float* __restrict__ acc) {
  __shared__ __align__(32) _Float16 zbuf[16 * LD_X];
  __shared__ __align__(32) _Float16 abuf[16 * LD_H];
  __shared__ float hd[16 * LD_S];
  __shared__ float px[16 * LD_PX];
  __shared__ float red[256];

  const int tid = threadIdx.x;
  const int lane = tid & 31;
  const int w = tid >> 5;
  const size_t row0 = (size_t)blockIdx.x * 16;

  for (int i = tid; i < 16 * 32; i += 256) {
    int r = i >> 5, c = i & 31;
    zbuf[r * LD_X + c] = (_Float16)((c < 16) ? zs[(row0 + r) * 16 + c] : 0.f);
  }
  __syncthreads();

  // a = relu(z @ act + act_b): 8 tiles / 8 waves
  {
    v8f a = {};
    a = wmma_f16(load_a_frag(zbuf, LD_X, lane), load_b_fr(ActF, w, lane), a);
    int n = lane & 15, mb = (lane >> 4) << 3;
#pragma unroll
    for (int v = 0; v < 8; ++v)
      abuf[(mb + v) * LD_H + w * 16 + n] = (_Float16)fmaxf(a[v] + act_b[w * 16 + n], 0.f);
  }
  __syncthreads();

  if (w == 0) {  // fused land|shot|move head
    v8f h = {};
#pragma unroll
    for (int kk = 0; kk < 4; ++kk)
      h = wmma_f16(load_a_frag(abuf + kk * 32, LD_H, lane), load_b_fr(HeadF, kk, lane), h);
    int n = lane & 15, mb = (lane >> 4) << 3;
#pragma unroll
    for (int v = 0; v < 8; ++v) hd[(mb + v) * LD_S + n] = h[v];
  } else if (w == 1 || w == 2) {  // projection to state space
    v8f p = {};
    int n0 = (w - 1) * 16;
    p = wmma_f16(load_a_frag(zbuf, LD_X, lane), load_b_fr(ProjF, w - 1, lane), p);
    int n = lane & 15, mb = (lane >> 4) << 3;
#pragma unroll
    for (int v = 0; v < 8; ++v) px[(mb + v) * LD_PX + n0 + n] = p[v] + proj_b[n0 + n];
  }
  __syncthreads();

  float land = 0.f, move = 0.f, picked = 0.f, maskc = 0.f, lpx = 0.f;
  const float noise = noise_std_p[0];
  const float LOG2PI = 1.8378770664093453f;

  for (int i = tid; i < 512; i += 256) {
    int r = i >> 5, d = i & 31;
    const float* x = xs + (row0 + r) * 23;
    float sv;
    if (d < 12)      sv = x[d];
    else if (d < 20) sv = shot_emb[((int)x[12]) * 8 + (d - 12)];
    else if (d < 24) sv = x[13 + (d - 20)];
    else             sv = player_emb[((int)x[17]) * 8 + (d - 24)];
    float df = (sv - px[r * LD_PX + d]) / noise;
    lpx += -0.5f * df * df - __logf(noise) - 0.5f * LOG2PI;
  }

  if (tid < 16) {
    int r = tid;
    const float* x = xs + (row0 + r) * 23;
    float l0 = hd[r * LD_S + 0] - x[18], l1 = hd[r * LD_S + 1] - x[19];
    land = l0 * l0 + l1 * l1;
    float m0 = hd[r * LD_S + 14] - x[21], m1_ = hd[r * LD_S + 15] - x[22];
    move = m0 * m0 + m1_ * m1_;
    int sid = (int)x[20];
    float mx = -1e30f;
    for (int j = 0; j < 12; ++j) mx = fmaxf(mx, hd[r * LD_S + 2 + j]);
    float se = 0.f;
    for (int j = 0; j < 12; ++j) se += __expf(hd[r * LD_S + 2 + j] - mx);
    float lp = hd[r * LD_S + 2 + sid] - mx - __logf(se);
    if (sid != 0) { picked = lp; maskc = 1.f; }
  }

  float vals[5] = {lpx, land, move, picked, maskc};
  const int accIdx[5] = {ACC_LPX, ACC_LAND, ACC_MOVE, ACC_PICK, ACC_MASK};
  for (int q = 0; q < 5; ++q) {
    __syncthreads();
    red[tid] = vals[q];
    __syncthreads();
    for (int s = 128; s > 0; s >>= 1) {
      if (tid < s) red[tid] += red[tid + s];
      __syncthreads();
    }
    if (tid == 0) atomicAdd(&acc[accIdx[q]], red[0]);
  }
}

__global__ void k_final(const float* __restrict__ acc, float* __restrict__ out) {
  if (threadIdx.x == 0 && blockIdx.x == 0) {
    float log_pxs = acc[ACC_LPX] / (float)kB;
    float land = acc[ACC_LAND] / (float)((size_t)kT * kB * 2);
    float move = acc[ACC_MOVE] / (float)((size_t)kT * kB * 2);
    float shot = -acc[ACC_PICK] / fmaxf(acc[ACC_MASK], 1.f);
    out[0] = log_pxs;
    out[1] = acc[ACC_KL] / (float)kB + acc[ACC_LOGR] / (float)kB + land + shot + move;
  }
}

// ---------------- launch ----------------

extern "C" void kernel_launch(void* const* d_in, const int* in_sizes, int n_in,
                              void* d_out, int out_size, void* d_ws, size_t ws_size,
                              hipStream_t stream) {
  const float* xs         = (const float*)d_in[1];
  const float* ts         = (const float*)d_in[2];
  const float* noise_std  = (const float*)d_in[3];
  const float* z0_noise   = (const float*)d_in[4];
  const float* bm_noise   = (const float*)d_in[5];
  const float* shot_emb   = (const float*)d_in[6];
  const float* player_emb = (const float*)d_in[7];
  const float* gru_wih    = (const float*)d_in[8];
  const float* gru_whh    = (const float*)d_in[9];
  const float* gru_bih    = (const float*)d_in[10];
  const float* gru_bhh    = (const float*)d_in[11];
  const float* enc_w      = (const float*)d_in[12];
  const float* enc_b      = (const float*)d_in[13];
  const float* qz0_w      = (const float*)d_in[14];
  const float* qz0_b      = (const float*)d_in[15];
  const float* f_w1 = (const float*)d_in[16]; const float* f_b1 = (const float*)d_in[17];
  const float* f_w2 = (const float*)d_in[18]; const float* f_b2 = (const float*)d_in[19];
  const float* f_w3 = (const float*)d_in[20]; const float* f_b3 = (const float*)d_in[21];
  const float* h_w1 = (const float*)d_in[22]; const float* h_b1 = (const float*)d_in[23];
  const float* h_w2 = (const float*)d_in[24]; const float* h_b2 = (const float*)d_in[25];
  const float* h_w3 = (const float*)d_in[26]; const float* h_b3 = (const float*)d_in[27];
  const float* g_w1 = (const float*)d_in[28]; const float* g_b1 = (const float*)d_in[29];
  const float* g_w2 = (const float*)d_in[30]; const float* g_b2 = (const float*)d_in[31];
  const float* proj_w = (const float*)d_in[32]; const float* proj_b = (const float*)d_in[33];
  const float* pz0_mean = (const float*)d_in[34]; const float* pz0_logstd = (const float*)d_in[35];
  const float* act_w = (const float*)d_in[36]; const float* act_b = (const float*)d_in[37];
  const float* act_land_w = (const float*)d_in[38];
  const float* act_shot_w = (const float*)d_in[39];
  const float* act_move_w = (const float*)d_in[40];

  char* ws = (char*)d_ws;
  size_t off = 0;
  auto take = [&](size_t bytes) -> char* {
    char* p = ws + off;
    off += (bytes + 255) & ~(size_t)255;
    return p;
  };
  const size_t TB = (size_t)kT * kB;
  _Float16* st16  = (_Float16*)take(TB * 32 * 2);
  _Float16* ctx16 = (_Float16*)take(TB * 64 * 2);
  float*    zsbuf = (float*)take(TB * 16 * 4);
  _Float16* WihF  = (_Float16*)take((size_t)32 * 384 * 2);
  _Float16* WhhF  = (_Float16*)take((size_t)128 * 384 * 2);
  _Float16* EncF  = (_Float16*)take((size_t)128 * 64 * 2);
  _Float16* F1F   = (_Float16*)take((size_t)96 * 128 * 2);
  _Float16* F2F   = (_Float16*)take((size_t)128 * 128 * 2);
  _Float16* F3F   = (_Float16*)take((size_t)128 * 16 * 2);
  _Float16* H1F   = (_Float16*)take((size_t)32 * 128 * 2);
  _Float16* H2F   = (_Float16*)take((size_t)128 * 128 * 2);
  _Float16* H3F   = (_Float16*)take((size_t)128 * 16 * 2);
  _Float16* ActF  = (_Float16*)take((size_t)32 * 128 * 2);
  _Float16* ProjF = (_Float16*)take((size_t)32 * 32 * 2);
  _Float16* HeadF = (_Float16*)take((size_t)128 * 16 * 2);
  float*    accb  = (float*)take(ACC_N * 4);

  k_zero_acc<<<1, 32, 0, stream>>>(accb);

  auto fr = [&](const float* W, _Float16* F, int n, int kdim, int kpad) {
    int total = (kpad >> 5) * (n >> 4) * 512;
    k_fragify<<<(total + 255) / 256, 256, 0, stream>>>(W, F, n, kdim, kpad);
  };
  fr(gru_wih, WihF, 384, 32, 32);
  fr(gru_whh, WhhF, 384, 128, 128);
  fr(enc_w,   EncF, 64, 128, 128);
  fr(f_w1,    F1F, 128, 80, 96);
  fr(f_w2,    F2F, 128, 128, 128);
  fr(f_w3,    F3F, 16, 128, 128);
  fr(h_w1,    H1F, 128, 16, 32);
  fr(h_w2,    H2F, 128, 128, 128);
  fr(h_w3,    H3F, 16, 128, 128);
  fr(act_w,   ActF, 128, 16, 32);
  fr(proj_w,  ProjF, 32, 16, 32);
  k_build_head_fr<<<(4 * 512 + 255) / 256, 256, 0, stream>>>(act_land_w, act_shot_w,
                                                             act_move_w, HeadF);

  {
    size_t total = TB * 32;
    k_embed<<<(unsigned)((total + 255) / 256), 256, 0, stream>>>(xs, shot_emb, player_emb, st16);
  }

  k_gru<<<kB / 16, 256, 0, stream>>>(st16, WihF, WhhF, EncF, gru_bih, gru_bhh, enc_b, ctx16);

  k_qz0<<<kB / 256, 256, 0, stream>>>(ctx16, qz0_w, qz0_b, z0_noise, pz0_mean, pz0_logstd,
                                      zsbuf, accb);

  k_sde<<<kB / 16, 256, 0, stream>>>(ctx16, ts, bm_noise,
                                     F1F, f_b1, F2F, f_b2, F3F, f_b3,
                                     H1F, h_b1, H2F, h_b2, H3F, h_b3,
                                     g_w1, g_b1, g_w2, g_b2, zsbuf, accb);

  k_decode<<<(unsigned)(TB / 16), 256, 0, stream>>>(zsbuf, xs, shot_emb, player_emb,
                                                    ActF, act_b, HeadF, ProjF, proj_b,
                                                    noise_std, accb);

  k_final<<<1, 32, 0, stream>>>(accb, (float*)d_out);
}
